// CausalSelfAttention_45337674776686
// MI455X (gfx1250) — matrix operand order repaired and checked by reading, not compiled, not run on hardware
//
#include <hip/hip_runtime.h>
#include <hip/hip_bf16.h>

// ---------------------------------------------------------------------------
// Causal self-attention for MI455X (gfx1250, wave32, WMMA).
// Matmuls on v_wmma_f32_16x16x32_bf16; tile staging via async global->LDS
// (GLOBAL_LOAD_ASYNC_TO_LDS_B128 + s_wait_asynccnt) with double buffering.
// ---------------------------------------------------------------------------

typedef __attribute__((ext_vector_type(16))) __bf16 v16bf;
typedef __attribute__((ext_vector_type(8)))  __bf16 v8bf;
typedef __attribute__((ext_vector_type(8)))  float  v8f;

constexpr int Bn = 4, Tt = 2048, Cc = 1024, Hh = 16, Dd = 64;

// ---- WMMA fragment helpers (layouts per CDNA5 ISA §7.12.2) -----

// A-matrix 16x32 bf16, tile row-major with row stride `ld` (elements).
// lane m = lane&15 ; kb = 8*(lane>>4); VGPR0-3 = K kb..kb+7, VGPR4-7 = K kb+16..kb+23
__device__ __forceinline__ v16bf ld_a(const __bf16* base, int ld) {
  int lane = threadIdx.x & 31;
  const __bf16* p = base + (lane & 15) * ld + ((lane >> 4) << 3);
  v8bf lo = *(const v8bf*)p;
  v8bf hi = *(const v8bf*)(p + 16);
  return __builtin_shufflevector(lo, hi, 0,1,2,3,4,5,6,7,8,9,10,11,12,13,14,15);
}

// B-matrix 32x16 bf16 where B[k][n] = tile[n][k] (tile row-major over n, stride ld).
// lane n = lane&15 ; kb = 8*(lane>>4); VGPR0-3 = K kb..kb+7, VGPR4-7 = K kb+16..kb+23
__device__ __forceinline__ v16bf ld_b(const __bf16* base, int ld) {
  int lane = threadIdx.x & 31;
  const __bf16* p = base + (lane & 15) * ld + ((lane >> 4) << 3);
  v8bf lo = *(const v8bf*)p;
  v8bf hi = *(const v8bf*)(p + 16);
  return __builtin_shufflevector(lo, hi, 0,1,2,3,4,5,6,7,8,9,10,11,12,13,14,15);
}

__device__ __forceinline__ v8f wmma_bf16(v16bf a, v16bf b, v8f c) {
  return __builtin_amdgcn_wmma_f32_16x16x32_bf16(false, a, false, b, (short)0, c,
                                                 false, false);
}

__device__ __forceinline__ v8f vzero8() {
  v8f z = {0.f, 0.f, 0.f, 0.f, 0.f, 0.f, 0.f, 0.f};
  return z;
}

// Async global->LDS 16B copy (per lane). LDS offset = low 32 bits of the
// generic pointer (AMDGPU LDS->generic cast is {aperture_hi, lds_offset};
// the flat->LDS mapping uses addr[31:0]).
__device__ __forceinline__ void async_copy_b128(void* lds_ptr, const void* gptr) {
  unsigned lds = (unsigned)(uintptr_t)lds_ptr;
  asm volatile("global_load_async_to_lds_b128 %0, %1, off"
               :: "v"(lds), "v"(gptr) : "memory");
}

// ---------------------------------------------------------------------------
// fp32 -> bf16 conversion
// ---------------------------------------------------------------------------
__global__ void cvt_bf16(const float* __restrict__ in, __bf16* __restrict__ out,
                         int n) {
  int i = blockIdx.x * blockDim.x + threadIdx.x;
  int stride = gridDim.x * blockDim.x;
  for (; i < n; i += stride) out[i] = (__bf16)in[i];
}

// ---------------------------------------------------------------------------
// GEMM: out[M,N] = A[M,K] @ W[N,K]^T   (both read row-major along K)
// 128x128 tile per 256-thread workgroup; 8 waves of 64(M)x32(N); K-step 32.
// Double-buffered LDS filled with async global->LDS copies.
// ---------------------------------------------------------------------------
template <typename OutT>
__global__ void __launch_bounds__(256)
gemm_xwt(const __bf16* __restrict__ A, const __bf16* __restrict__ W,
         OutT* __restrict__ out, int M, int N, int K) {
  constexpr int LDT = 40;  // 32 + 8 pad (80B rows, 16B multiple)
  __shared__ __align__(16) __bf16 As[2][128 * LDT];
  __shared__ __align__(16) __bf16 Bs[2][128 * LDT];

  const int n0 = blockIdx.x * 128;
  const int m0 = blockIdx.y * 128;
  const int tid = threadIdx.x;
  const int wave = tid >> 5;
  const int wm = (wave >> 2) * 64;  // 0 / 64
  const int wn = (wave & 3) * 32;   // 0..96

  // Issue the async fill of one 128x32 A tile + 128x32 W tile into buf.
  // 2 chunks/thread for each matrix -> 4 async instructions per wave per tile.
  auto issue_tile = [&](int k0, int buf) {
    for (int c = tid; c < 512; c += 256) {
      int row = c >> 2, kc = (c & 3) << 3;
      async_copy_b128(&As[buf][row * LDT + kc],
                      &A[(size_t)(m0 + row) * K + k0 + kc]);
      async_copy_b128(&Bs[buf][row * LDT + kc],
                      &W[(size_t)(n0 + row) * K + k0 + kc]);
    }
  };

  v8f acc[4][2];
#pragma unroll
  for (int i = 0; i < 4; ++i)
#pragma unroll
    for (int j = 0; j < 2; ++j) acc[i][j] = vzero8();

  const int nk = K / 32;
  issue_tile(0, 0);
  for (int t = 0; t < nk; ++t) {
    const int buf = t & 1;
    if (t + 1 < nk) {
      // Prefetch next tile into the other buffer while we compute this one.
      // (Safe: end-of-iteration barrier below guarantees nobody still reads it.)
      issue_tile((t + 1) * 32, buf ^ 1);
      // Async loads complete in order: <=4 outstanding => this tile's 4 done.
      asm volatile("s_wait_asynccnt 0x4" ::: "memory");
    } else {
      asm volatile("s_wait_asynccnt 0x0" ::: "memory");
    }
    __syncthreads();  // all waves' portions of tile t are in LDS

    v16bf b0 = ld_b(&Bs[buf][(wn + 0) * LDT], LDT);
    v16bf b1 = ld_b(&Bs[buf][(wn + 16) * LDT], LDT);
#pragma unroll
    for (int i = 0; i < 4; ++i) {
      v16bf a = ld_a(&As[buf][(wm + i * 16) * LDT], LDT);
      acc[i][0] = wmma_bf16(a, b0, acc[i][0]);
      acc[i][1] = wmma_bf16(a, b1, acc[i][1]);
    }
    __syncthreads();  // done reading buf before iter t+1 overwrites buf^1's pair
  }

  // C/D layout: VGPR r, lane: N = lane&15, M = r + 8*(lane>>4)
  const int lane = tid & 31;
  const int nc = lane & 15, mb = (lane >> 4) * 8;
#pragma unroll
  for (int i = 0; i < 4; ++i)
#pragma unroll
    for (int j = 0; j < 2; ++j) {
      OutT* o = out + (size_t)(m0 + wm + i * 16 + mb) * N + (n0 + wn + j * 16 + nc);
#pragma unroll
      for (int r = 0; r < 8; ++r) o[(size_t)r * N] = (OutT)acc[i][j][r];
    }
}

// ---------------------------------------------------------------------------
// Flash attention (causal): grid (T/64, H, B), 128 threads (4 waves).
// Wave w owns query rows q0+16w .. q0+16w+15; streams 32-key blocks.
// K tile staged via async global->LDS; V transposed through VGPRs.
// ---------------------------------------------------------------------------
__global__ void __launch_bounds__(128)
flash_attn(const __bf16* __restrict__ Q, const __bf16* __restrict__ Kg,
           const __bf16* __restrict__ V, __bf16* __restrict__ Y) {
  constexpr int LDK = 72;  // 64 + 8 pad
  constexpr int LDV = 40;  // 32 + 8 pad
  constexpr int LDP = 40;
  __shared__ __align__(16) __bf16 Ks[32 * LDK];       // K block [32 keys][64 d]
  __shared__ __align__(16) __bf16 Vt[64 * LDV];       // V^T     [64 d][32 keys]
  __shared__ __align__(16) __bf16 Ps[4 * 16 * LDP];   // per-wave P [16 q][32 k]

  const int q0 = blockIdx.x * 64;
  const int h = blockIdx.y, b = blockIdx.z;
  const int tid = threadIdx.x, lane = tid & 31, w = tid >> 5;
  const size_t base = ((size_t)b * Tt) * Cc + (size_t)h * Dd;
  const __bf16* qp = Q + base;
  const __bf16* kp = Kg + base;
  const __bf16* vp = V + base;
  __bf16* yp = Y + base;

  const int qb = q0 + w * 16;
  const int ncol = lane & 15, hi8 = (lane >> 4) * 8;
  const float scale = 0.125f;  // 1/sqrt(64)

  // Q fragments (A-layout), resident for the whole kernel: d = 0..31, 32..63
  v16bf aq0 = ld_a(qp + (size_t)qb * Cc, Cc);
  v16bf aq1 = ld_a(qp + (size_t)qb * Cc + 32, Cc);

  float mi[8], li[8];
  v8f acc[4];
#pragma unroll
  for (int r = 0; r < 8; ++r) { mi[r] = -1e30f; li[r] = 0.f; }
#pragma unroll
  for (int j = 0; j < 4; ++j) acc[j] = vzero8();

  const int nblk = (q0 + 64) / 32;
  for (int kt = 0; kt < nblk; ++kt) {
    __syncthreads();  // previous block's K/V reads complete before refill
    const int kr0 = kt * 32;

    // K tile: 32 rows x 64 d, async global->LDS (2 x 16B chunks per thread)
    for (int c = tid; c < 256; c += 128) {
      int row = c >> 3, dc = (c & 7) << 3;
      async_copy_b128(&Ks[row * LDK + dc], &kp[(size_t)(kr0 + row) * Cc + dc]);
    }
    // V tile transposed through VGPRs: Vt[d][k]
    {
      int kk = tid & 31, d0 = (tid >> 5) * 16;
      const __bf16* vr = &vp[(size_t)(kr0 + kk) * Cc + d0];
      v8bf va = *(const v8bf*)vr;
      v8bf vb2 = *(const v8bf*)(vr + 8);
#pragma unroll
      for (int i = 0; i < 8; ++i) {
        Vt[(d0 + i) * LDV + kk] = va[i];
        Vt[(d0 + 8 + i) * LDV + kk] = vb2[i];
      }
    }
    // Hint the next block toward L2 while we compute this one.
    if (kt + 1 < nblk) {
      __builtin_prefetch(&kp[(size_t)(kr0 + 32) * Cc], 0, 1);
      __builtin_prefetch(&vp[(size_t)(kr0 + 32) * Cc], 0, 1);
    }
    asm volatile("s_wait_asynccnt 0x0" ::: "memory");
    __syncthreads();

    if (kr0 <= qb + 15) {  // wave-uniform: block touches this wave's rows
      // S = Q K^T  (two 16x16 tiles over 32 keys; contract d=64 in 2 steps)
      v8f s0 = vzero8(), s1 = vzero8();
      s0 = wmma_bf16(aq0, ld_b(&Ks[0 * LDK + 0], LDK), s0);
      s0 = wmma_bf16(aq1, ld_b(&Ks[0 * LDK + 32], LDK), s0);
      s1 = wmma_bf16(aq0, ld_b(&Ks[16 * LDK + 0], LDK), s1);
      s1 = wmma_bf16(aq1, ld_b(&Ks[16 * LDK + 32], LDK), s1);

      __bf16* pw = &Ps[w * 16 * LDP];
#pragma unroll
      for (int r = 0; r < 8; ++r) {
        const int qg = qb + hi8 + r;
        float v0 = s0[r] * scale;
        if (kr0 + ncol > qg) v0 = -1e30f;
        float v1 = s1[r] * scale;
        if (kr0 + 16 + ncol > qg) v1 = -1e30f;
        // row max across the 16 lanes holding this row's columns
        float t = fmaxf(v0, v1);
        t = fmaxf(t, __shfl_xor(t, 1, 32));
        t = fmaxf(t, __shfl_xor(t, 2, 32));
        t = fmaxf(t, __shfl_xor(t, 4, 32));
        t = fmaxf(t, __shfl_xor(t, 8, 32));
        const float mnew = fmaxf(mi[r], t);
        const float al = __expf(mi[r] - mnew);
        const float p0 = __expf(v0 - mnew);
        const float p1 = __expf(v1 - mnew);
        float rs = p0 + p1;
        rs += __shfl_xor(rs, 1, 32);
        rs += __shfl_xor(rs, 2, 32);
        rs += __shfl_xor(rs, 4, 32);
        rs += __shfl_xor(rs, 8, 32);
        li[r] = li[r] * al + rs;
        mi[r] = mnew;
        acc[0][r] *= al; acc[1][r] *= al; acc[2][r] *= al; acc[3][r] *= al;
        // C-layout -> A-layout via LDS: P[m][k]
        pw[(hi8 + r) * LDP + ncol] = (__bf16)p0;
        pw[(hi8 + r) * LDP + ncol + 16] = (__bf16)p1;
      }
      asm volatile("s_wait_dscnt 0" ::: "memory");  // P visible before reload
      v16bf ap = ld_a(pw, LDP);
#pragma unroll
      for (int j = 0; j < 4; ++j)
        acc[j] = wmma_bf16(ap, ld_b(&Vt[j * 16 * LDV], LDV), acc[j]);
    }
  }

  // epilogue: normalize and store y (bf16) back in [b*T + t][h*64 + d] layout
#pragma unroll
  for (int r = 0; r < 8; ++r) {
    const float inv = 1.0f / li[r];
    const int qg = qb + hi8 + r;
#pragma unroll
    for (int j = 0; j < 4; ++j)
      yp[(size_t)qg * Cc + j * 16 + ncol] = (__bf16)(acc[j][r] * inv);
  }
}

// ---------------------------------------------------------------------------
// Host launch
// ---------------------------------------------------------------------------
extern "C" void kernel_launch(void* const* d_in, const int* in_sizes, int n_in,
                              void* d_out, int out_size, void* d_ws, size_t ws_size,
                              hipStream_t stream) {
  const float* x  = (const float*)d_in[0];
  const float* Wq = (const float*)d_in[1];
  const float* Wk = (const float*)d_in[2];
  const float* Wv = (const float*)d_in[3];
  const float* Wo = (const float*)d_in[4];
  float* out = (float*)d_out;

  const int M = Bn * Tt;               // 8192
  const size_t NX = (size_t)M * Cc;    // 8,388,608
  const size_t NW = (size_t)Cc * Cc;   // 1,048,576

  // workspace layout (bf16): x | Wq Wk Wv Wo | Q | K | V | Y  (~88 MB)
  __bf16* xb  = (__bf16*)d_ws;
  __bf16* wqb = xb + NX;
  __bf16* wkb = wqb + NW;
  __bf16* wvb = wkb + NW;
  __bf16* wob = wvb + NW;
  __bf16* qb  = wob + NW;
  __bf16* kb  = qb + NX;
  __bf16* vb  = kb + NX;
  __bf16* yb  = vb + NX;

  cvt_bf16<<<2048, 256, 0, stream>>>(x, xb, (int)NX);
  cvt_bf16<<<512, 256, 0, stream>>>(Wq, wqb, (int)NW);
  cvt_bf16<<<512, 256, 0, stream>>>(Wk, wkb, (int)NW);
  cvt_bf16<<<512, 256, 0, stream>>>(Wv, wvb, (int)NW);
  cvt_bf16<<<512, 256, 0, stream>>>(Wo, wob, (int)NW);

  dim3 gg(Cc / 128, M / 128);  // (8, 64)
  gemm_xwt<__bf16><<<gg, 256, 0, stream>>>(xb, wqb, qb, M, Cc, Cc);
  gemm_xwt<__bf16><<<gg, 256, 0, stream>>>(xb, wkb, kb, M, Cc, Cc);
  gemm_xwt<__bf16><<<gg, 256, 0, stream>>>(xb, wvb, vb, M, Cc, Cc);

  flash_attn<<<dim3(Tt / 64, Hh, Bn), 128, 0, stream>>>(qb, kb, vb, yb);

  gemm_xwt<float><<<gg, 256, 0, stream>>>(yb, wob, out, M, Cc, Cc);
}
